// SegFormerMultiheadAttention_13099650252907
// MI455X (gfx1250) — matrix operand order repaired and checked by reading, not compiled, not run on hardware
//
#include <hip/hip_runtime.h>
#include <stdint.h>

// ---------------------------------------------------------------------------
// SegFormer multi-head attention (B=4, N=4096, C=512, H=8, dh=64, SR=4)
// MI455X / gfx1250: bf16 WMMA + TDM tile DMA (GEMM) + async LDS DMA (attn).
// ---------------------------------------------------------------------------

typedef __bf16 v16bf __attribute__((ext_vector_type(16)));
typedef float  v8f   __attribute__((ext_vector_type(8)));
typedef unsigned int u32x4 __attribute__((ext_vector_type(4)));
typedef int i32x8 __attribute__((ext_vector_type(8)));
typedef int i32x4 __attribute__((ext_vector_type(4)));

union FragBF { v16bf v; uint4 q[2]; };

__device__ __forceinline__ v8f wmma_bf16(v16bf a, v16bf b, v8f c) {
  return __builtin_amdgcn_wmma_f32_16x16x32_bf16(
      false, a, false, b, (short)0, c, false, false);
}

__device__ __forceinline__ uint16_t f2bf(float f) {
  union { float f; uint32_t u; } c; c.f = f;
  uint32_t u = c.u;
  uint32_t r = u + 0x7FFFu + ((u >> 16) & 1u);   // round-to-nearest-even
  return (uint16_t)(r >> 16);
}

__device__ __forceinline__ uint32_t lds_addr(const void* p) {
  return (uint32_t)(uintptr_t)p;   // generic LDS addr: low 32 bits = LDS offset
}

// Async 16-byte global -> LDS copy (ASYNCcnt-tracked; CDNA5 §15.18.3 op 98).
__device__ __forceinline__ void async_copy16(uint32_t lds_off, const void* gptr) {
  asm volatile("global_load_async_to_lds_b128 %0, %1, off"
               :: "v"(lds_off), "v"((uint64_t)(uintptr_t)gptr) : "memory");
}
__device__ __forceinline__ void wait_async0() {
  asm volatile("s_wait_asynccnt 0x0" ::: "memory");
}

// ---------------------------------------------------------------------------
// TDM: one descriptor moves a 2D tile (tl1 rows x tl0 elems, row stride
// stride0 elems, 2-byte elements) global -> LDS, with LDS padding of
// 4 DWORDs (16B) after every 16 DWORDs (64B) => padded row stride of 40
// bf16 elements, matching ASTR/BSTR below.  CDNA5 ISA.
// This toolchain's builtin takes 6 args (g0, g1, g2, g3, g-extra, cpol).
// ---------------------------------------------------------------------------
__device__ __forceinline__ void tdm_load_2d(uint32_t lds_off, const void* gptr,
                                            uint32_t td0, uint32_t td1,
                                            uint32_t tl0, uint32_t tl1,
                                            uint32_t stride0) {
  uint64_t ga = (uint64_t)(uintptr_t)gptr;
  u32x4 g0;
  g0.x = 1u;                                   // count=1, user descriptor
  g0.y = lds_off;                              // lds_addr (bytes)
  g0.z = (uint32_t)ga;                         // global_addr[31:0]
  g0.w = (uint32_t)(ga >> 32) | (2u << 30);    // global_addr[56:32] | type=2
  i32x8 g1;
  // data_size=1 (2B), pad_enable, pad_interval=3 (16 DW), pad_amount=3 (4 DW)
  g1[0] = (int)((1u << 16) | (1u << 20) | (3u << 22) | (3u << 25));
  g1[1] = (int)((td0 & 0xFFFFu) << 16);                     // tensor_dim0 lo
  g1[2] = (int)((td0 >> 16) | ((td1 & 0xFFFFu) << 16));     // dim0 hi | dim1 lo
  g1[3] = (int)((td1 >> 16) | (tl0 << 16));                 // dim1 hi | tile_dim0
  g1[4] = (int)tl1;                                         // tile_dim1 (dim2=0)
  g1[5] = (int)stride0;                                     // tensor_dim0_stride
  g1[6] = 0;
  g1[7] = 0;
  i32x4 z4 = {0, 0, 0, 0};
  i32x8 z8 = {0, 0, 0, 0, 0, 0, 0, 0};
  __builtin_amdgcn_tensor_load_to_lds(g0, g1, z4, z4, z8, 0);
}

// ---------------------------------------------------------------------------
// f32 -> bf16 convert, 8 elements / thread
// ---------------------------------------------------------------------------
__global__ __launch_bounds__(256)
void cvt_bf16_kernel(const float* __restrict__ in, uint16_t* __restrict__ out, int n8) {
  int i = blockIdx.x * 256 + threadIdx.x;
  if (i >= n8) return;
  const float4* p = reinterpret_cast<const float4*>(in) + (size_t)i * 2;
  float4 a = p[0], b = p[1];
  uint16_t t[8];
  t[0] = f2bf(a.x); t[1] = f2bf(a.y); t[2] = f2bf(a.z); t[3] = f2bf(a.w);
  t[4] = f2bf(b.x); t[5] = f2bf(b.y); t[6] = f2bf(b.z); t[7] = f2bf(b.w);
  *reinterpret_cast<uint4*>(out + (size_t)i * 8) = *reinterpret_cast<uint4*>(t);
}

// ---------------------------------------------------------------------------
// Transposing convert: W[K,N] f32 -> Wt[N,K] bf16 (one-time per weight).
// ---------------------------------------------------------------------------
__global__ __launch_bounds__(256)
void cvt_bf16_t_kernel(const float* __restrict__ in, uint16_t* __restrict__ out,
                       int K, int N, int total8) {
  int i = blockIdx.x * 256 + threadIdx.x;
  if (i >= total8) return;
  int kc = K >> 3;
  int n  = i / kc;
  int k0 = (i - n * kc) * 8;
  uint16_t t[8];
  #pragma unroll
  for (int j = 0; j < 8; ++j)
    t[j] = f2bf(in[(size_t)(k0 + j) * N + n]);
  *reinterpret_cast<uint4*>(out + (size_t)n * K + k0) = *reinterpret_cast<uint4*>(t);
}

// ---------------------------------------------------------------------------
// im2col for the SR=4 conv
// ---------------------------------------------------------------------------
__global__ __launch_bounds__(256)
void im2col_kernel(const float* __restrict__ x, uint16_t* __restrict__ p, int n8) {
  int i = blockIdx.x * 256 + threadIdx.x;
  if (i >= n8) return;
  size_t idx = (size_t)i * 8;
  int row = (int)(idx >> 13);
  int kk  = (int)(idx & 8191);
  int b   = row >> 8;
  int pp  = row & 255;
  int oy  = pp >> 4, ox = pp & 15;
  int kh  = kk >> 11;
  int kw  = (kk >> 9) & 3;
  int ci  = kk & 511;
  int n   = (oy * 4 + kh) * 64 + ox * 4 + kw;
  const float4* s4 = reinterpret_cast<const float4*>(x + ((size_t)b * 4096 + n) * 512 + ci);
  float4 a = s4[0], c = s4[1];
  uint16_t t[8];
  t[0] = f2bf(a.x); t[1] = f2bf(a.y); t[2] = f2bf(a.z); t[3] = f2bf(a.w);
  t[4] = f2bf(c.x); t[5] = f2bf(c.y); t[6] = f2bf(c.z); t[7] = f2bf(c.w);
  *reinterpret_cast<uint4*>(p + idx) = *reinterpret_cast<uint4*>(t);
}

// ---------------------------------------------------------------------------
// V head-transpose: vt[(b*8+h)][d][kv] = v[(b*256+kv)*512 + h*64 + d]
// ---------------------------------------------------------------------------
__global__ __launch_bounds__(256)
void vtrans_kernel(const uint16_t* __restrict__ v, uint16_t* __restrict__ vt, int total) {
  int i = blockIdx.x * 256 + threadIdx.x;
  if (i >= total) return;                 // 65536 (8 kv each)
  int kv8 = i & 31;
  int d   = (i >> 5) & 63;
  int bh  = i >> 11;
  int b = bh >> 3, h = bh & 7;
  int kv0 = kv8 * 8;
  uint16_t t[8];
  #pragma unroll
  for (int j = 0; j < 8; ++j)
    t[j] = v[((size_t)(b * 256 + kv0 + j)) * 512 + h * 64 + d];
  *reinterpret_cast<uint4*>(vt + (size_t)bh * 16384 + (size_t)d * 256 + kv0) =
      *reinterpret_cast<uint4*>(t);
}

// ---------------------------------------------------------------------------
// Double-buffered TDM WMMA GEMM:
//   C[M,N] = A[M,K](bf16 row-major) * Bt[N,K](bf16 pre-transposed) + bias
// Wave 0 issues one TDM descriptor per tile (A: 128x32, Bt: 64x32) into the
// ping-pong LDS buffers; everyone waits TENSORcnt==0 + barrier, then WMMA.
// ---------------------------------------------------------------------------
#define BM 128
#define BN 64
#define BK 32
#define ASTR 40   // BK + 8 pad = TDM pad (16B per 64B row)
#define BSTR 40

__global__ __launch_bounds__(256)
void gemm_bf16_kernel(const uint16_t* __restrict__ A, const uint16_t* __restrict__ Bt,
                      const float* __restrict__ bias,
                      uint16_t* __restrict__ Obf, float* __restrict__ Of,
                      int M, int N, int K) {
  __shared__ alignas(16) uint16_t As[2][BM * ASTR];
  __shared__ alignas(16) uint16_t Bs[2][BN * BSTR];   // [n][k] (Bt rows)

  const int tid  = threadIdx.x;
  const int lane = tid & 31;
  const int wid  = tid >> 5;
  const int wm   = wid >> 1;
  const int wn   = wid & 1;
  const int m0   = blockIdx.y * BM;
  const int n0   = blockIdx.x * BN;
  const int lm   = lane & 15;
  const int lh   = lane >> 4;

  auto issue_tile = [&](int kt, int buf) {
    int k0 = kt * BK;
    tdm_load_2d(lds_addr(&As[buf][0]), &A[(size_t)m0 * K + k0],
                (uint32_t)K, (uint32_t)M, BK, BM, (uint32_t)K);
    tdm_load_2d(lds_addr(&Bs[buf][0]), &Bt[(size_t)n0 * K + k0],
                (uint32_t)K, (uint32_t)N, BK, BN, (uint32_t)K);
  };

  v8f acc[2][2];
  acc[0][0] = v8f{}; acc[0][1] = v8f{}; acc[1][0] = v8f{}; acc[1][1] = v8f{};

  const int T = K / BK;
  if (wid == 0) issue_tile(0, 0);

  for (int kt = 0; kt < T; ++kt) {
    const int cur = kt & 1;
    __builtin_amdgcn_s_wait_tensorcnt(0);   // wave0: tile kt landed; others: no-op
    __syncthreads();                        // tile visible; old buffer free
    if (wid == 0 && kt + 1 < T) issue_tile(kt + 1, cur ^ 1);  // DMA overlaps math

    const uint16_t* as = As[cur];
    const uint16_t* bs = Bs[cur];
    FragBF a0, a1, b0, b1;
    a0.q[0] = *reinterpret_cast<const uint4*>(&as[(wm * 32 + lm) * ASTR + lh * 8]);
    a0.q[1] = *reinterpret_cast<const uint4*>(&as[(wm * 32 + lm) * ASTR + 16 + lh * 8]);
    a1.q[0] = *reinterpret_cast<const uint4*>(&as[(wm * 32 + 16 + lm) * ASTR + lh * 8]);
    a1.q[1] = *reinterpret_cast<const uint4*>(&as[(wm * 32 + 16 + lm) * ASTR + 16 + lh * 8]);
    b0.q[0] = *reinterpret_cast<const uint4*>(&bs[(wn * 32 + lm) * BSTR + lh * 8]);
    b0.q[1] = *reinterpret_cast<const uint4*>(&bs[(wn * 32 + lm) * BSTR + 16 + lh * 8]);
    b1.q[0] = *reinterpret_cast<const uint4*>(&bs[(wn * 32 + 16 + lm) * BSTR + lh * 8]);
    b1.q[1] = *reinterpret_cast<const uint4*>(&bs[(wn * 32 + 16 + lm) * BSTR + 16 + lh * 8]);

    acc[0][0] = wmma_bf16(a0.v, b0.v, acc[0][0]);
    acc[0][1] = wmma_bf16(a0.v, b1.v, acc[0][1]);
    acc[1][0] = wmma_bf16(a1.v, b0.v, acc[1][0]);
    acc[1][1] = wmma_bf16(a1.v, b1.v, acc[1][1]);
  }

  #pragma unroll
  for (int ti = 0; ti < 2; ++ti)
    #pragma unroll
    for (int tj = 0; tj < 2; ++tj) {
      int ng = n0 + wn * 32 + tj * 16 + lm;
      float bb = bias ? bias[ng] : 0.0f;
      #pragma unroll
      for (int r = 0; r < 8; ++r) {
        int mg = m0 + wm * 32 + ti * 16 + r + 8 * lh;
        float val = acc[ti][tj][r] + bb;
        if (Of) Of[(size_t)mg * N + ng] = val;
        else    Obf[(size_t)mg * N + ng] = f2bf(val);
      }
    }
}

// ---------------------------------------------------------------------------
// LayerNorm over C=512, one block per row, bf16 output.
// ---------------------------------------------------------------------------
__global__ __launch_bounds__(256)
void ln_kernel(const float* __restrict__ xr, const float* __restrict__ gamma,
               const float* __restrict__ beta, uint16_t* __restrict__ out) {
  __shared__ float sh_s[8], sh_ss[8];
  int row = blockIdx.x;
  const float* px = xr + (size_t)row * 512;
  float s = 0.f, ss = 0.f;
  for (int i = threadIdx.x; i < 512; i += 256) { float v = px[i]; s += v; ss += v * v; }
  #pragma unroll
  for (int m = 16; m >= 1; m >>= 1) { s += __shfl_xor(s, m, 32); ss += __shfl_xor(ss, m, 32); }
  if ((threadIdx.x & 31) == 0) { sh_s[threadIdx.x >> 5] = s; sh_ss[threadIdx.x >> 5] = ss; }
  __syncthreads();
  float ts = 0.f, tss = 0.f;
  #pragma unroll
  for (int w = 0; w < 8; ++w) { ts += sh_s[w]; tss += sh_ss[w]; }
  float mu   = ts * (1.0f / 512.0f);
  float var  = tss * (1.0f / 512.0f) - mu * mu;
  float rstd = rsqrtf(var + 1e-5f);
  for (int i = threadIdx.x; i < 512; i += 256) {
    float v = (px[i] - mu) * rstd * gamma[i] + beta[i];
    out[(size_t)row * 512 + i] = f2bf(v);
  }
}

// ---------------------------------------------------------------------------
// Fused attention (async-DMA K and Vt slabs, 64 WMMAs / wave).
// ---------------------------------------------------------------------------
#define KSTR 72    // dh(64)+8
#define VSTR 264   // nkv(256)+8
#define PSTR 264

__global__ __launch_bounds__(256)
void attn_kernel(const uint16_t* __restrict__ Q, const uint16_t* __restrict__ Kp,
                 const uint16_t* __restrict__ Vt_g, uint16_t* __restrict__ O) {
  extern __shared__ uint16_t smem[];
  uint16_t* Ks = smem;                       // [256][KSTR]
  uint16_t* Vt = Ks + 256 * KSTR;            // [64][VSTR]
  uint16_t* Ps = Vt + 64 * VSTR;             // 8 x [16][PSTR]

  const int bid = blockIdx.x;
  const int qb  = bid & 31;
  const int bh  = bid >> 5;
  const int b   = bh >> 3;
  const int h   = bh & 7;

  const int tid = threadIdx.x, lane = tid & 31, wid = tid >> 5;
  const int lm = lane & 15, lh = lane >> 4;

  {
    const uint16_t* src = Kp + ((size_t)(b * 256 + tid) * 512 + h * 64);
    #pragma unroll
    for (int j = 0; j < 8; ++j)
      async_copy16(lds_addr(&Ks[tid * KSTR + j * 8]), src + j * 8);
  }
  {
    const uint16_t* src = Vt_g + (size_t)bh * 16384;
    #pragma unroll
    for (int j = 0; j < 8; ++j) {
      int chunk = tid + j * 256;
      int d = chunk >> 5, seg = chunk & 31;
      async_copy16(lds_addr(&Vt[d * VSTR + seg * 8]), src + d * 256 + seg * 8);
    }
  }
  wait_async0();
  __syncthreads();

  const int qrow0 = qb * 128 + wid * 16;
  const size_t qbase = ((size_t)(b * 4096 + qrow0)) * 512 + h * 64;

  FragBF aq[2];
  #pragma unroll
  for (int c = 0; c < 2; ++c) {
    aq[c].q[0] = *reinterpret_cast<const uint4*>(&Q[qbase + (size_t)lm * 512 + c * 32 + lh * 8]);
    aq[c].q[1] = *reinterpret_cast<const uint4*>(&Q[qbase + (size_t)lm * 512 + c * 32 + 16 + lh * 8]);
  }

  v8f S[16];
  #pragma unroll
  for (int t = 0; t < 16; ++t) {
    v8f a = v8f{};
    #pragma unroll
    for (int c = 0; c < 2; ++c) {
      FragBF bk;
      const uint16_t* kr = &Ks[(t * 16 + lm) * KSTR + c * 32 + lh * 8];
      bk.q[0] = *reinterpret_cast<const uint4*>(kr);
      bk.q[1] = *reinterpret_cast<const uint4*>(kr + 16);
      a = wmma_bf16(aq[c].v, bk.v, a);
    }
    S[t] = a;
  }

  float inv[8];
  const float sl2e = 0.125f * 1.44269504088896f;
  #pragma unroll
  for (int r = 0; r < 8; ++r) {
    float mx = -1e30f;
    #pragma unroll
    for (int t = 0; t < 16; ++t) mx = fmaxf(mx, S[t][r]);
    #pragma unroll
    for (int m = 8; m >= 1; m >>= 1) mx = fmaxf(mx, __shfl_xor(mx, m, 32));
    float sum = 0.f;
    #pragma unroll
    for (int t = 0; t < 16; ++t) {
      float p = exp2f((S[t][r] - mx) * sl2e);
      S[t][r] = p;
      sum += p;
    }
    #pragma unroll
    for (int m = 8; m >= 1; m >>= 1) sum += __shfl_xor(sum, m, 32);
    inv[r] = 1.0f / sum;
  }

  uint16_t* Pw = Ps + wid * 16 * PSTR;
  #pragma unroll
  for (int t = 0; t < 16; ++t)
    #pragma unroll
    for (int r = 0; r < 8; ++r)
      Pw[(r + 8 * lh) * PSTR + t * 16 + lm] = f2bf(S[t][r]);

  #pragma unroll
  for (int dt = 0; dt < 4; ++dt) {
    v8f a = v8f{};
    #pragma unroll
    for (int ck = 0; ck < 8; ++ck) {
      FragBF ap, bv;
      const uint16_t* pr = &Pw[lm * PSTR + ck * 32 + lh * 8];
      ap.q[0] = *reinterpret_cast<const uint4*>(pr);
      ap.q[1] = *reinterpret_cast<const uint4*>(pr + 16);
      const uint16_t* vr = &Vt[(dt * 16 + lm) * VSTR + ck * 32 + lh * 8];
      bv.q[0] = *reinterpret_cast<const uint4*>(vr);
      bv.q[1] = *reinterpret_cast<const uint4*>(vr + 16);
      a = wmma_bf16(ap.v, bv.v, a);
    }
    #pragma unroll
    for (int r = 0; r < 8; ++r) {
      int m = r + 8 * lh;
      size_t off = ((size_t)(b * 4096 + qrow0 + m)) * 512 + h * 64 + dt * 16 + lm;
      O[off] = f2bf(a[r] * inv[r]);
    }
  }
}

// ---------------------------------------------------------------------------
// Host side
// ---------------------------------------------------------------------------
extern "C" void kernel_launch(void* const* d_in, const int* in_sizes, int n_in,
                              void* d_out, int out_size, void* d_ws, size_t ws_size,
                              hipStream_t stream) {
  (void)in_sizes; (void)n_in; (void)out_size; (void)ws_size;

  const float* x     = (const float*)d_in[0];
  const float* wq    = (const float*)d_in[1];
  const float* bq    = (const float*)d_in[2];
  const float* wk    = (const float*)d_in[3];
  const float* bk    = (const float*)d_in[4];
  const float* wv    = (const float*)d_in[5];
  const float* bv    = (const float*)d_in[6];
  const float* w_sr  = (const float*)d_in[7];
  const float* b_sr  = (const float*)d_in[8];
  const float* gamma = (const float*)d_in[9];
  const float* beta  = (const float*)d_in[10];
  const float* wp    = (const float*)d_in[11];
  const float* bp    = (const float*)d_in[12];
  float* out = (float*)d_out;

  uint8_t* ws = (uint8_t*)d_ws;
  size_t off = 0;
  auto take = [&](size_t bytes) {
    uint8_t* p = ws + off;
    off += (bytes + 255) & ~(size_t)255;
    return p;
  };
  const size_t XN  = 4ull * 4096 * 512;
  const size_t WN  = 512ull * 512;
  const size_t SRN = 8192ull * 512;
  const size_t PN  = 1024ull * 8192;
  const size_t RN  = 1024ull * 512;

  uint16_t* x_bf    = (uint16_t*)take(XN * 2);
  uint16_t* wq_t    = (uint16_t*)take(WN * 2);
  uint16_t* wk_t    = (uint16_t*)take(WN * 2);
  uint16_t* wv_t    = (uint16_t*)take(WN * 2);
  uint16_t* wp_t    = (uint16_t*)take(WN * 2);
  uint16_t* wsr_t   = (uint16_t*)take(SRN * 2);
  uint16_t* patches = (uint16_t*)take(PN * 2);
  uint16_t* q_bf    = (uint16_t*)take(XN * 2);
  float*    xr_f32  = (float*)   take(RN * 4);
  uint16_t* xr_bf   = (uint16_t*)take(RN * 2);
  uint16_t* k_bf    = (uint16_t*)take(RN * 2);
  uint16_t* v_bf    = (uint16_t*)take(RN * 2);
  uint16_t* vt_bf   = (uint16_t*)take(RN * 2);
  uint16_t* ao_bf   = (uint16_t*)take(XN * 2);

  auto cvt = [&](const float* src, uint16_t* dst, size_t n) {
    int n8 = (int)(n / 8);
    cvt_bf16_kernel<<<(n8 + 255) / 256, 256, 0, stream>>>(src, dst, n8);
  };
  auto cvt_t = [&](const float* src, uint16_t* dst, int K, int N) {
    int t8 = (int)((size_t)K * N / 8);
    cvt_bf16_t_kernel<<<(t8 + 255) / 256, 256, 0, stream>>>(src, dst, K, N, t8);
  };

  cvt(x, x_bf, XN);
  cvt_t(wq, wq_t, 512, 512);
  cvt_t(wk, wk_t, 512, 512);
  cvt_t(wv, wv_t, 512, 512);
  cvt_t(wp, wp_t, 512, 512);
  cvt_t(w_sr, wsr_t, 8192, 512);

  {
    int n8 = (int)(PN / 8);
    im2col_kernel<<<(n8 + 255) / 256, 256, 0, stream>>>(x, patches, n8);
  }

  gemm_bf16_kernel<<<dim3(512 / BN, 16384 / BM), 256, 0, stream>>>(
      x_bf, wq_t, bq, q_bf, nullptr, 16384, 512, 512);

  gemm_bf16_kernel<<<dim3(512 / BN, 1024 / BM), 256, 0, stream>>>(
      patches, wsr_t, b_sr, nullptr, xr_f32, 1024, 512, 8192);

  ln_kernel<<<1024, 256, 0, stream>>>(xr_f32, gamma, beta, xr_bf);

  gemm_bf16_kernel<<<dim3(512 / BN, 1024 / BM), 256, 0, stream>>>(
      xr_bf, wk_t, bk, k_bf, nullptr, 1024, 512, 512);
  gemm_bf16_kernel<<<dim3(512 / BN, 1024 / BM), 256, 0, stream>>>(
      xr_bf, wv_t, bv, v_bf, nullptr, 1024, 512, 512);

  vtrans_kernel<<<65536 / 256, 256, 0, stream>>>(v_bf, vt_bf, 65536);

  {
    size_t smem = (size_t)(256 * KSTR + 64 * VSTR + 8 * 16 * PSTR) * sizeof(uint16_t);
    attn_kernel<<<1024, 256, smem, stream>>>(q_bf, k_bf, vt_bf, ao_bf);
  }

  gemm_bf16_kernel<<<dim3(512 / BN, 16384 / BM), 256, 0, stream>>>(
      ao_bf, wp_t, bp, nullptr, out, 16384, 512, 512);
}
